// GNNLayer_58093727645864
// MI455X (gfx1250) — matrix operand order repaired and checked by reading, not compiled, not run on hardware
//
#include <hip/hip_runtime.h>
#include <hip/hip_bf16.h>
#include <math.h>
#include <stdint.h>

// ---------------- problem constants ----------------
#define Bz    32
#define Nn    128
#define Mm    256
#define Ee    8192
#define HIDc  768
#define Hh    12
#define HDc   64
#define INTERc 3072
#define ROWS  (Bz * Nn)          // 4096 token rows

typedef __attribute__((ext_vector_type(16))) _Float16 v16h;
typedef __attribute__((ext_vector_type(4)))  _Float16 v4h;
typedef __attribute__((ext_vector_type(8)))  float    v8f;

#define WMMA_F16(a, b, c) \
  __builtin_amdgcn_wmma_f32_16x16x32_f16(false, (a), false, (b), (short)0, (c), false, false)

// async copy of one 16-byte chunk: global -> LDS (tracked by ASYNCcnt)
__device__ __forceinline__ void async_b128(unsigned lds_addr, const void* gaddr) {
  asm volatile("global_load_async_to_lds_b128 %0, %1, off"
               :: "v"(lds_addr), "v"(gaddr) : "memory");
}
__device__ __forceinline__ void async_wait0() {
  asm volatile("s_wait_asynccnt 0" ::: "memory");
}
__device__ __forceinline__ unsigned lds_off(const void* p) {
  return (unsigned)(uintptr_t)p;   // low 32 bits of generic LDS ptr = DS byte address
}

// ---- fragment loaders (contiguous k order, same in A and B; the ISA table order differs) ----
// A 16x32 f16 row-major in LDS, row stride `stride` halves.
// lane: m = lane&15, half = lane>>4; e=0..15 -> K=half*16+e (one contiguous run)
__device__ __forceinline__ v16h frag_a(const _Float16* base, int stride) {
  int lane = threadIdx.x & 31;
  int m = lane & 15, half = lane >> 4;
  const _Float16* p = base + m * stride + half * 16;
  v16h r;
#pragma unroll
  for (int e = 0; e < 16; ++e) r[e] = p[e];
  return r;
}
// B 32x16 f16 staged TRANSPOSED in LDS: Bt[n][k], row stride `stride` halves.
// lane: n = lane&15, half = lane>>4; e=0..15 -> K=half*16+e (one contiguous run)
__device__ __forceinline__ v16h frag_b(const _Float16* base, int stride) {
  int lane = threadIdx.x & 31;
  int n = lane & 15, half = lane >> 4;
  const _Float16* p = base + n * stride + half * 16;
  v16h r;
#pragma unroll
  for (int e = 0; e < 16; ++e) r[e] = p[e];
  return r;
}

// ---------------- generic tiled WMMA GEMM ----------------
// C[M x N] = A[M x K] @ W[K x N] + bias (+ GELU / residual); optional f16 output
enum { EPI_NONE = 0, EPI_GELU = 1, EPI_RESID = 2 };

template <int EPI, int OUTH>
__global__ void gemm_wmma_kernel(const float* __restrict__ A, const float* __restrict__ W,
                                 const float* __restrict__ bias, const float* __restrict__ resid,
                                 float* __restrict__ Cf, _Float16* __restrict__ C16,
                                 int Mdim, int Ndim, int Kdim) {
  __shared__ _Float16 Ash[64][40];   // [m][k]   (row stride 80B -> 8B aligned v4h slots)
  __shared__ _Float16 Bsh[64][40];   // transposed: [n][k]
  const int n0 = blockIdx.x * 64, m0 = blockIdx.y * 64;
  const int t = threadIdx.x;         // 128 (4 waves)
  const int wave = t >> 5, wr = wave >> 1, wc = wave & 1;

  v8f acc[2][2];
#pragma unroll
  for (int i = 0; i < 2; ++i)
#pragma unroll
    for (int j = 0; j < 2; ++j)
#pragma unroll
      for (int e = 0; e < 8; ++e) acc[i][j][e] = 0.0f;

#pragma unroll 1
  for (int k0 = 0; k0 < Kdim; k0 += 32) {
    // A tile 64x32: vectorized float4 loads -> packed 4xf16 LDS stores
#pragma unroll
    for (int it = 0; it < 4; ++it) {
      int idx = t + it * 128;                  // 512 float4 chunks
      int r = idx >> 3, cq = idx & 7;          // row, quad-column
      float4 f = *(const float4*)&A[(size_t)(m0 + r) * Kdim + k0 + cq * 4];
      v4h h = { (_Float16)f.x, (_Float16)f.y, (_Float16)f.z, (_Float16)f.w };
      *(v4h*)&Ash[r][cq * 4] = h;
    }
    // W tile 32x64 -> Bt[n][k]: float4 loads along n (coalesced), scalar transposed DS scatter
#pragma unroll
    for (int it = 0; it < 4; ++it) {
      int idx = t + it * 128;                  // 512 float4 chunks
      int kk = idx >> 4, nq = idx & 15;        // k-row, quad-n
      float4 f = *(const float4*)&W[(size_t)(k0 + kk) * Ndim + n0 + nq * 4];
      Bsh[nq * 4 + 0][kk] = (_Float16)f.x;
      Bsh[nq * 4 + 1][kk] = (_Float16)f.y;
      Bsh[nq * 4 + 2][kk] = (_Float16)f.z;
      Bsh[nq * 4 + 3][kk] = (_Float16)f.w;
    }
    __syncthreads();
    v16h af[2], bf[2];
    af[0] = frag_a(&Ash[wr * 32 + 0][0], 40);
    af[1] = frag_a(&Ash[wr * 32 + 16][0], 40);
    bf[0] = frag_b(&Bsh[wc * 32 + 0][0], 40);
    bf[1] = frag_b(&Bsh[wc * 32 + 16][0], 40);
#pragma unroll
    for (int i = 0; i < 2; ++i)
#pragma unroll
      for (int j = 0; j < 2; ++j) acc[i][j] = WMMA_F16(af[i], bf[j], acc[i][j]);
    __syncthreads();
  }

  const int lane = t & 31, cn = lane & 15, rh = lane >> 4;
#pragma unroll
  for (int i = 0; i < 2; ++i)
#pragma unroll
    for (int j = 0; j < 2; ++j) {
      int n = n0 + wc * 32 + j * 16 + cn;
      float bv = bias[n];
#pragma unroll
      for (int r = 0; r < 8; ++r) {
        int m = m0 + wr * 32 + i * 16 + r + 8 * rh;
        float v = acc[i][j][r] + bv;
        if (EPI == EPI_GELU) v = 0.5f * v * (1.0f + erff(v * 0.70710678118654752f));
        if (EPI == EPI_RESID) v += resid[(size_t)m * Ndim + n];
        if (OUTH) C16[(size_t)m * Ndim + n] = (_Float16)v;
        else      Cf [(size_t)m * Ndim + n] = v;
      }
    }
}

// ---------------- attention: q.k^T raw scores ----------------
// 256 threads = 8 waves; wave w owns i-tile w (16 rows x all 128 j).
// q/k tiles staged via async-to-LDS (raw f16 copy, 16B chunks, strided rows).
__global__ void attn_scores_kernel(const _Float16* __restrict__ q, const _Float16* __restrict__ k,
                                   float* __restrict__ scores) {
  const int bh = blockIdx.x, b = bh / Hh, h = bh % Hh;
  __shared__ _Float16 qsh[128][64];   // [i][d]
  __shared__ _Float16 ksh[128][64];   // Bt order: [j][d]
  const int t = threadIdx.x;          // 256
  const char* gq = (const char*)(q + (size_t)(b * Nn) * HIDc + h * HDc);
  const char* gk = (const char*)(k + (size_t)(b * Nn) * HIDc + h * HDc);
  const unsigned lq = lds_off(&qsh[0][0]), lk = lds_off(&ksh[0][0]);
  for (int idx = t; idx < 128 * 8; idx += 256) {    // 8 x 16B chunks per 128B row
    int r = idx >> 3, c = (idx & 7) * 16;
    async_b128(lq + r * 128 + c, gq + (size_t)r * (HIDc * 2) + c);
    async_b128(lk + r * 128 + c, gk + (size_t)r * (HIDc * 2) + c);
  }
  async_wait0();
  __syncthreads();

  const int wave = t >> 5, lane = t & 31, cn = lane & 15, rh = lane >> 4;
  const int itile = wave;
  v8f acc[8];
#pragma unroll
  for (int jt = 0; jt < 8; ++jt)
#pragma unroll
    for (int e = 0; e < 8; ++e) acc[jt][e] = 0.0f;

#pragma unroll 1
  for (int k0 = 0; k0 < 64; k0 += 32) {
    v16h af = frag_a(&qsh[itile * 16][k0], 64);
#pragma unroll
    for (int jt = 0; jt < 8; ++jt)
      acc[jt] = WMMA_F16(af, frag_b(&ksh[jt * 16][k0], 64), acc[jt]);
  }
#pragma unroll
  for (int jt = 0; jt < 8; ++jt)
#pragma unroll
    for (int r = 0; r < 8; ++r) {
      int i = itile * 16 + r + 8 * rh, j = jt * 16 + cn;
      scores[((size_t)bh * 128 + i) * 128 + j] = acc[jt][r];
    }
}

// ---------------- attention: edge-bias scores ----------------
// one wave per (b,i): C[16x128] = qslice(12->16 pad x 64) @ edge[b,i,:,:]^T; += into scores.
// 16KB edge slice + 12 q rows async-copied; only the 4 pad rows touched by VALU.
__global__ void edge_bias_scores_kernel(const _Float16* __restrict__ q,
                                        const _Float16* __restrict__ edge_emb,
                                        float* __restrict__ scores) {
  const int bi = blockIdx.x, b = bi >> 7, i = bi & 127;
  __shared__ _Float16 ash[16][64];    // rows = head (12 valid, 4 zero pad)
  __shared__ _Float16 esh[128][64];   // Bt order: [j][d] (memory layout already matches)
  const int t = threadIdx.x;          // 32
  const char* ge = (const char*)(edge_emb + (size_t)(b * Nn + i) * Nn * HDc);
  const char* gq = (const char*)(q + (size_t)(b * Nn + i) * HIDc);   // 12 heads x 128B, contiguous
  const unsigned le = lds_off(&esh[0][0]);
  const unsigned la = lds_off(&ash[0][0]);
  for (int c = t; c < 1024; c += 32)                 // 1024 x 16B = 16KB
    async_b128(le + c * 16, ge + (size_t)c * 16);
  for (int c = t; c < 12 * 8; c += 32)               // 12 rows x 8 chunks (q rows contiguous)
    async_b128(la + c * 16, gq + (size_t)c * 16);
  for (int idx = t; idx < 4 * 64; idx += 32) {       // zero pad rows 12..15
    ash[12 + (idx >> 6)][idx & 63] = (_Float16)0.0f;
  }
  async_wait0();
  __syncthreads();

  v16h af0 = frag_a(&ash[0][0], 64);
  v16h af1 = frag_a(&ash[0][32], 64);
  const int lane = t & 31, cn = lane & 15, rh = lane >> 4;
#pragma unroll 1
  for (int jt = 0; jt < 8; ++jt) {
    v8f acc;
#pragma unroll
    for (int e = 0; e < 8; ++e) acc[e] = 0.0f;
    acc = WMMA_F16(af0, frag_b(&esh[jt * 16][0], 64), acc);
    acc = WMMA_F16(af1, frag_b(&esh[jt * 16][32], 64), acc);
#pragma unroll
    for (int r = 0; r < 8; ++r) {
      int hr = r + 8 * rh;
      if (hr < Hh) {
        int j = jt * 16 + cn;
        scores[(((size_t)b * Hh + hr) * 128 + i) * 128 + j] += acc[r];
      }
    }
  }
}

// ---------------- masked softmax: f32 scores in, f16 probs out; scale 1/8 here ----------------
__global__ void softmax_kernel(const float* __restrict__ scores, const float* __restrict__ node_mask,
                               _Float16* __restrict__ probs) {
  const int row = blockIdx.x;                 // b*H*128 + h*128 + i
  const int b = row / (Hh * 128);
  const int t = threadIdx.x;                  // 128
  __shared__ float red[128];
  float v = scores[(size_t)row * 128 + t] * 0.125f + (node_mask[b * Nn + t] - 1.0f) * 1e9f;
  red[t] = v; __syncthreads();
  for (int s = 64; s > 0; s >>= 1) { if (t < s) red[t] = fmaxf(red[t], red[t + s]); __syncthreads(); }
  float mx = red[0]; __syncthreads();
  float e = __expf(v - mx);
  red[t] = e; __syncthreads();
  for (int s = 64; s > 0; s >>= 1) { if (t < s) red[t] += red[t + s]; __syncthreads(); }
  probs[(size_t)row * 128 + t] = (_Float16)(e / red[0]);
}

// ---------------- attention: probs @ v -> ctx ----------------
// 256 threads = 8 waves; wave w owns i-tile w. probs tile (32KB contiguous) async-copied;
// v tile staged transposed ([d][j]) by VALU while the async copy is in flight.
__global__ void attn_ctx_kernel(const _Float16* __restrict__ probs, const _Float16* __restrict__ v,
                                float* __restrict__ ctx) {
  const int bh = blockIdx.x, b = bh / Hh, h = bh % Hh;
  __shared__ _Float16 psh[128][128];  // [i][j]
  __shared__ _Float16 vsh[64][128];   // Bt order: [d][j]
  const int t = threadIdx.x;          // 256
  const char* gp = (const char*)(probs + (size_t)bh * 128 * 128);
  const unsigned lp = lds_off(&psh[0][0]);
  for (int c = t; c < 2048; c += 256)                // 2048 x 16B = 32KB
    async_b128(lp + c * 16, gp + (size_t)c * 16);
  for (int idx = t; idx < 64 * 128; idx += 256) {
    int d = idx >> 7, j = idx & 127;
    vsh[d][j] = v[(size_t)(b * Nn + j) * HIDc + h * HDc + d];
  }
  async_wait0();
  __syncthreads();

  const int wave = t >> 5, lane = t & 31, cn = lane & 15, rh = lane >> 4;
  const int itile = wave;
  v8f acc[4];
#pragma unroll
  for (int nt = 0; nt < 4; ++nt)
#pragma unroll
    for (int e = 0; e < 8; ++e) acc[nt][e] = 0.0f;

#pragma unroll 1
  for (int k0 = 0; k0 < 128; k0 += 32) {
    v16h af = frag_a(&psh[itile * 16][k0], 128);
#pragma unroll
    for (int nt = 0; nt < 4; ++nt)
      acc[nt] = WMMA_F16(af, frag_b(&vsh[nt * 16][k0], 128), acc[nt]);
  }
#pragma unroll
  for (int nt = 0; nt < 4; ++nt)
#pragma unroll
    for (int r = 0; r < 8; ++r) {
      int i = itile * 16 + r + 8 * rh, d = nt * 16 + cn;
      ctx[(size_t)(b * Nn + i) * HIDc + h * HDc + d] = acc[nt][r];
    }
}

// ---------------- LayerNorm over 768-wide rows ----------------
__global__ void layernorm768_kernel(const float* __restrict__ x, const float* __restrict__ g,
                                    const float* __restrict__ bta, float* __restrict__ y) {
  const int row = blockIdx.x, t = threadIdx.x;   // 256 threads
  __shared__ float red[256];
  const float* xr = x + (size_t)row * HIDc;
  float s = 0.0f;
  for (int d = t; d < HIDc; d += 256) s += xr[d];
  red[t] = s; __syncthreads();
  for (int st = 128; st > 0; st >>= 1) { if (t < st) red[t] += red[t + st]; __syncthreads(); }
  float mean = red[0] / (float)HIDc; __syncthreads();
  float s2 = 0.0f;
  for (int d = t; d < HIDc; d += 256) { float dd = xr[d] - mean; s2 += dd * dd; }
  red[t] = s2; __syncthreads();
  for (int st = 128; st > 0; st >>= 1) { if (t < st) red[t] += red[t + st]; __syncthreads(); }
  float inv = rsqrtf(red[0] / (float)HIDc + 1e-12f);
  for (int d = t; d < HIDc; d += 256)
    y[(size_t)row * HIDc + d] = (xr[d] - mean) * inv * g[d] + bta[d];
}

// ---------------- bond embedding: cate lookup + float linear, each LayerNorm(64) ----------------
__global__ void bond_embed_kernel(const int* __restrict__ cate, const float* __restrict__ flo,
                                  const float* __restrict__ emb_cate, const float* __restrict__ w_float,
                                  const float* __restrict__ b_float,
                                  const float* __restrict__ g_c, const float* __restrict__ b_c,
                                  const float* __restrict__ g_f, const float* __restrict__ b_f,
                                  float* __restrict__ hidden_bond) {
  const int starts[4] = {0, 8, 16, 20};
  const int bm = blockIdx.x, d = threadIdx.x;   // 64 threads
  __shared__ float red[64];
  float sc = 0.0f;
#pragma unroll
  for (int j = 0; j < 4; ++j) { int idx = cate[bm * 4 + j] + starts[j]; sc += emb_cate[idx * 64 + d]; }
  float sf = b_float[d];
#pragma unroll
  for (int j = 0; j < 4; ++j) sf += flo[bm * 4 + j] * w_float[j * 64 + d];

  red[d] = sc; __syncthreads();
  for (int s = 32; s > 0; s >>= 1) { if (d < s) red[d] += red[d + s]; __syncthreads(); }
  float mc = red[0] * (1.0f / 64.0f); __syncthreads();
  float dc = sc - mc; red[d] = dc * dc; __syncthreads();
  for (int s = 32; s > 0; s >>= 1) { if (d < s) red[d] += red[d + s]; __syncthreads(); }
  float outc = dc * rsqrtf(red[0] * (1.0f / 64.0f) + 1e-12f) * g_c[d] + b_c[d]; __syncthreads();

  red[d] = sf; __syncthreads();
  for (int s = 32; s > 0; s >>= 1) { if (d < s) red[d] += red[d + s]; __syncthreads(); }
  float mf = red[0] * (1.0f / 64.0f); __syncthreads();
  float df = sf - mf; red[d] = df * df; __syncthreads();
  for (int s = 32; s > 0; s >>= 1) { if (d < s) red[d] += red[d + s]; __syncthreads(); }
  float outf = df * rsqrtf(red[0] * (1.0f / 64.0f) + 1e-12f) * g_f[d] + b_f[d];

  hidden_bond[(size_t)bm * 64 + d] = outc + outf;
}

// ---------------- edge tensor (f16): zero / scatter / path+ring bias ----------------
__global__ void zero_kernel(float* __restrict__ p, size_t nwords) {
  size_t i = (size_t)blockIdx.x * blockDim.x + threadIdx.x;
  size_t stride = (size_t)gridDim.x * blockDim.x;
  for (; i < nwords; i += stride) p[i] = 0.0f;
}

__global__ void scatter_edges_kernel(const int* __restrict__ ei, const int* __restrict__ ea,
                                     const float* __restrict__ hb, _Float16* __restrict__ edge_emb) {
  const int e = blockIdx.x, d = threadIdx.x;   // 64 threads
  int b = ei[e], i = ei[Ee + e], j = ei[2 * Ee + e];
  int bb = ea[e], m = ea[Ee + e];
  edge_emb[(((size_t)b * Nn + i) * Nn + j) * HDc + d] = (_Float16)hb[((size_t)bb * Mm + m) * HDc + d];
}

__global__ void path_ring_bias_kernel(const int* __restrict__ spl, const int* __restrict__ src,
                                      const float* __restrict__ emb_path, const float* __restrict__ g_p,
                                      const float* __restrict__ b_p,
                                      const float* __restrict__ emb_ring, const float* __restrict__ g_r,
                                      const float* __restrict__ b_r,
                                      _Float16* __restrict__ edge_emb) {
  const int lin = blockIdx.x;                 // b*127*127 + i*127 + j
  const int b = lin / (127 * 127);
  const int rem = lin - b * (127 * 127);
  const int i = rem / 127, j = rem - (rem / 127) * 127;
  const int d = threadIdx.x;                  // 64 threads
  __shared__ float red[64];
  int pi = spl[lin], ri = src[lin];
  float pv = (pi == 0) ? 0.0f : emb_path[pi * 64 + d];   // padding_idx=0 row zeroed
  float rv = (ri == 0) ? 0.0f : emb_ring[ri * 64 + d];

  red[d] = pv; __syncthreads();
  for (int s = 32; s > 0; s >>= 1) { if (d < s) red[d] += red[d + s]; __syncthreads(); }
  float mp = red[0] * (1.0f / 64.0f); __syncthreads();
  float dp = pv - mp; red[d] = dp * dp; __syncthreads();
  for (int s = 32; s > 0; s >>= 1) { if (d < s) red[d] += red[d + s]; __syncthreads(); }
  float pb = dp * rsqrtf(red[0] * (1.0f / 64.0f) + 1e-12f) * g_p[d] + b_p[d]; __syncthreads();

  red[d] = rv; __syncthreads();
  for (int s = 32; s > 0; s >>= 1) { if (d < s) red[d] += red[d + s]; __syncthreads(); }
  float mr = red[0] * (1.0f / 64.0f); __syncthreads();
  float dr = rv - mr; red[d] = dr * dr; __syncthreads();
  for (int s = 32; s > 0; s >>= 1) { if (d < s) red[d] += red[d + s]; __syncthreads(); }
  float rb = dr * rsqrtf(red[0] * (1.0f / 64.0f) + 1e-12f) * g_r[d] + b_r[d];

  size_t idx = (((size_t)b * Nn + (i + 1)) * Nn + (j + 1)) * HDc + d;
  edge_emb[idx] = (_Float16)((float)edge_emb[idx] + pb + rb);
}

// ---------------- host-side orchestration ----------------
extern "C" void kernel_launch(void* const* d_in, const int* in_sizes, int n_in,
                              void* d_out, int out_size, void* d_ws, size_t ws_size,
                              hipStream_t stream) {
  (void)in_sizes; (void)n_in; (void)out_size; (void)ws_size;
  // inputs in setup_inputs() dict order
  const float* hidden_node    = (const float*)d_in[0];
  const float* node_mask      = (const float*)d_in[1];
  const int*   bond_feat_cate = (const int*)d_in[2];
  const float* bond_feat_flt  = (const float*)d_in[3];
  const int*   edge_index     = (const int*)d_in[4];
  const int*   edge_bond      = (const int*)d_in[5];
  const int*   spl            = (const int*)d_in[6];
  const int*   src            = (const int*)d_in[7];
  /* d_in[8] = paths2, unused */
  const float* emb_path   = (const float*)d_in[9];
  const float* emb_ring   = (const float*)d_in[10];
  const float* emb_cate   = (const float*)d_in[11];
  const float* w_float    = (const float*)d_in[12];
  const float* b_float    = (const float*)d_in[13];
  const float* ln_path_g  = (const float*)d_in[14];
  const float* ln_path_b  = (const float*)d_in[15];
  const float* ln_ring_g  = (const float*)d_in[16];
  const float* ln_ring_b  = (const float*)d_in[17];
  const float* ln_cate_g  = (const float*)d_in[18];
  const float* ln_cate_b  = (const float*)d_in[19];
  const float* ln_float_g = (const float*)d_in[20];
  const float* ln_float_b = (const float*)d_in[21];
  const float* wq = (const float*)d_in[22]; const float* bq = (const float*)d_in[23];
  const float* wk = (const float*)d_in[24]; const float* bk = (const float*)d_in[25];
  const float* wv = (const float*)d_in[26]; const float* bv = (const float*)d_in[27];
  const float* wo = (const float*)d_in[28]; const float* bo = (const float*)d_in[29];
  const float* ln_attn_g = (const float*)d_in[30]; const float* ln_attn_b = (const float*)d_in[31];
  const float* wi = (const float*)d_in[32]; const float* bi = (const float*)d_in[33];
  const float* w_out = (const float*)d_in[34]; const float* b_out = (const float*)d_in[35];
  const float* ln_out_g = (const float*)d_in[36]; const float* ln_out_b = (const float*)d_in[37];

  float* out = (float*)d_out;

  // workspace carve-out (256B-aligned chunks)
  char* wsb = (char*)d_ws;
  auto carve = [&](size_t bytes) -> char* {
    char* p = wsb; wsb += (bytes + 255) & ~(size_t)255; return p;
  };
  const size_t EDGE_H = (size_t)Bz * Nn * Nn * HDc;                 // 33,554,432 halves
  _Float16* edge16 = (_Float16*)carve(EDGE_H * 2);
  float*    hb     = (float*)   carve((size_t)Bz * Mm * HDc * 4);
  _Float16* q16    = (_Float16*)carve((size_t)ROWS * HIDc * 2);
  _Float16* k16    = (_Float16*)carve((size_t)ROWS * HIDc * 2);
  _Float16* v16    = (_Float16*)carve((size_t)ROWS * HIDc * 2);
  float*    scores = (float*)   carve((size_t)Bz * Hh * Nn * Nn * 4);
  _Float16* probs  = (_Float16*)carve((size_t)Bz * Hh * Nn * Nn * 2);
  float*    ctx    = (float*)   carve((size_t)ROWS * HIDc * 4);
  float*    inter  = (float*)   carve((size_t)ROWS * INTERc * 4);
  float*    t0     = (float*)   carve((size_t)ROWS * HIDc * 4);  // ctx@wo + bo + residual
  float*    t1     = (float*)   carve((size_t)ROWS * HIDc * 4);  // attn_out = LN(t0)
  float*    t2     = (float*)   carve((size_t)ROWS * HIDc * 4);  // ffn pre-LN

  // 1) bond embedding
  bond_embed_kernel<<<Bz * Mm, 64, 0, stream>>>(bond_feat_cate, bond_feat_flt, emb_cate, w_float,
                                                b_float, ln_cate_g, ln_cate_b, ln_float_g, ln_float_b, hb);
  // 2) dense edge tensor (f16): zero -> scatter -> path/ring bias
  zero_kernel<<<4096, 256, 0, stream>>>((float*)edge16, EDGE_H / 2);
  scatter_edges_kernel<<<Ee, 64, 0, stream>>>(edge_index, edge_bond, hb, edge16);
  path_ring_bias_kernel<<<Bz * 127 * 127, 64, 0, stream>>>(spl, src, emb_path, ln_path_g, ln_path_b,
                                                           emb_ring, ln_ring_g, ln_ring_b, edge16);
  // 3) QKV projections (WMMA GEMMs, f16 outputs)
  dim3 g768(HIDc / 64, ROWS / 64);
  gemm_wmma_kernel<EPI_NONE, 1><<<g768, 128, 0, stream>>>(hidden_node, wq, bq, nullptr, nullptr, q16, ROWS, HIDc, HIDc);
  gemm_wmma_kernel<EPI_NONE, 1><<<g768, 128, 0, stream>>>(hidden_node, wk, bk, nullptr, nullptr, k16, ROWS, HIDc, HIDc);
  gemm_wmma_kernel<EPI_NONE, 1><<<g768, 128, 0, stream>>>(hidden_node, wv, bv, nullptr, nullptr, v16, ROWS, HIDc, HIDc);
  // 4) attention scores = q.k^T, then += q.edge^T, then masked softmax -> f16 probs
  attn_scores_kernel<<<Bz * Hh, 256, 0, stream>>>(q16, k16, scores);
  edge_bias_scores_kernel<<<Bz * Nn, 32, 0, stream>>>(q16, edge16, scores);
  softmax_kernel<<<Bz * Hh * Nn, 128, 0, stream>>>(scores, node_mask, probs);
  // 5) context = probs @ v
  attn_ctx_kernel<<<Bz * Hh, 256, 0, stream>>>(probs, v16, ctx);
  // 6) output projection + residual, LN
  gemm_wmma_kernel<EPI_RESID, 0><<<g768, 128, 0, stream>>>(ctx, wo, bo, hidden_node, t0, nullptr, ROWS, HIDc, HIDc);
  layernorm768_kernel<<<ROWS, 256, 0, stream>>>(t0, ln_attn_g, ln_attn_b, t1);
  // 7) FFN: GELU GEMM then projection + residual, LN into d_out
  dim3 g3072(INTERc / 64, ROWS / 64);
  gemm_wmma_kernel<EPI_GELU, 0><<<g3072, 128, 0, stream>>>(t1, wi, bi, nullptr, inter, nullptr, ROWS, INTERc, HIDc);
  gemm_wmma_kernel<EPI_RESID, 0><<<g768, 128, 0, stream>>>(inter, w_out, b_out, t1, t2, nullptr, ROWS, HIDc, INTERc);
  layernorm768_kernel<<<ROWS, 256, 0, stream>>>(t2, ln_out_g, ln_out_b, out);
}
